// TransformerBlock_17231408792466
// MI455X (gfx1250) — matrix operand order repaired and checked by reading, compiled, not run on hardware
//
#include <hip/hip_runtime.h>
#include <hip/hip_bf16.h>

// ---------------------------------------------------------------------------
// Transformer block for MI455X (gfx1250): f16 WMMA (f32 accum) everywhere,
// flash-attention online softmax (never materializes [B,H,S,S] scores),
// async global->LDS staging (ASYNCcnt) for GEMM A-tiles where available.
// ---------------------------------------------------------------------------

typedef __attribute__((ext_vector_type(16))) _Float16 v16h;
typedef __attribute__((ext_vector_type(8)))  _Float16 v8h;
typedef __attribute__((ext_vector_type(8)))  float    v8f;

// The async-to-LDS builtin is declared as (v4i AS1*, v4i AS3*, imm, imm)
// with GCC-style vector pointees (clang prints AS1 as __device__ in HIP).
typedef int tb_v4i __attribute__((vector_size(16)));
typedef __attribute__((address_space(1))) tb_v4i as1_v4i;
typedef __attribute__((address_space(3))) tb_v4i as3_v4i;

#define TB_B      2
#define TB_S      2048
#define TB_D      512
#define TB_H      8
#define TB_DH     64
#define TB_DFF    2048
#define TB_M      (TB_B * TB_S)   // 4096 token rows

// gfx1250 async global->LDS copy path (ASYNCcnt). Guarded so the file always
// compiles; fallback is a plain VGPR-bounce copy.
#if __has_builtin(__builtin_amdgcn_global_load_async_to_lds_b128) && \
    __has_builtin(__builtin_amdgcn_s_wait_asynccnt)
#define TB_ASYNC 1
#else
#define TB_ASYNC 0
#endif

__device__ __forceinline__ void lds_fill16(_Float16* l, const _Float16* g) {
#if TB_ASYNC
    __builtin_amdgcn_global_load_async_to_lds_b128(
        (as1_v4i*)g, (as3_v4i*)l, 0, 0);
#else
    *(v8h*)l = *(const v8h*)g;
#endif
}
__device__ __forceinline__ void async_wait_le1() {
#if TB_ASYNC
    __builtin_amdgcn_s_wait_asynccnt(1);
#endif
}
__device__ __forceinline__ void async_wait_le0() {
#if TB_ASYNC
    __builtin_amdgcn_s_wait_asynccnt(0);
#endif
}

// ---------------------------------------------------------------------------
// fp32 -> fp16 convert (one element per thread)
// ---------------------------------------------------------------------------
__global__ void cvt_f32_to_f16(const float* __restrict__ in,
                               _Float16* __restrict__ out, int n) {
    int i = blockIdx.x * blockDim.x + threadIdx.x;
    if (i < n) out[i] = (_Float16)in[i];
}

// W is [K x N] row-major; WT is [N x K] row-major (f16).
__global__ void transpose_cvt(const float* __restrict__ W,
                              _Float16* __restrict__ WT, int K, int N) {
    int i = blockIdx.x * blockDim.x + threadIdx.x;
    if (i < K * N) {
        int n = i / K;
        int k = i - n * K;
        WT[i] = (_Float16)W[(size_t)k * N + n];
    }
}

// ---------------------------------------------------------------------------
// Generic WMMA GEMM:  out[M x N] = A[M x K] @ BT[N x K]^T + bias
// Block = 128 threads = 4 waves. Block tile 32(M) x 256(N); each wave owns
// 32 x 64 via 8 f32 accumulators (2 A-frags x 4 B-frags per k-step).
// The block-shared 32x32 A chunk is staged in LDS with double-buffered
// async global->LDS copies; B fragments stream directly from global
// (contiguous v16h thanks to the pre-transposed weights).
//
// Fragment layouts used here (wave32; contiguous k order chosen by this code, not the ISA table order):
//   A (16x32 f16): lane&15 = row M; lane>=16 selects K-halves;
//                  half-elems i -> K = k0 + hi*16 + i (contiguous)
//   B (32x16 f16): lane&15 = col N; elems i -> K = k0 + hi*16 + i (contiguous)
//   C/D (16x16 f32): lane&15 = N; VGPR v -> M = v + (hi ? 8 : 0)
// ---------------------------------------------------------------------------
__global__ __launch_bounds__(128) void gemm_wmma(
    const _Float16* __restrict__ A, const _Float16* __restrict__ BT,
    const float* __restrict__ bias, float* __restrict__ outF,
    _Float16* __restrict__ outH, int Mdim, int Ndim, int Kdim,
    int relu, int vtrans) {
    __shared__ alignas(16) _Float16 sA[2][32][32];

    const int tid  = threadIdx.x;
    const int lane = tid & 31;
    const int wave = tid >> 5;
    const int nlo  = lane & 15;
    const int hi   = (lane >> 4) & 1;
    const int m0   = blockIdx.y * 32;
    const int n0   = blockIdx.x * 256 + wave * 64;

    // cooperative A staging: each of the 128 threads owns one 16B segment
    const int arow = tid >> 2;        // 0..31
    const int aseg = (tid & 3) * 8;   // half-offset 0,8,16,24
    const _Float16* agp = A + (size_t)(m0 + arow) * Kdim + aseg;

    v8f acc[2][4] = {};

    lds_fill16(&sA[0][arow][aseg], agp);   // prologue fill, k0 = 0

    for (int k0 = 0; k0 < Kdim; k0 += 32) {
        const int buf = (k0 >> 5) & 1;
        if (k0 + 32 < Kdim) {
            lds_fill16(&sA[buf ^ 1][arow][aseg], agp + k0 + 32);
            async_wait_le1();              // fill for current buf complete
        } else {
            async_wait_le0();
        }
        __syncthreads();

        v16h aF[2];
#pragma unroll
        for (int u = 0; u < 2; ++u) {
            const _Float16* ap = &sA[buf][u * 16 + nlo][0];
            *(v8h*)&aF[u]       = *(const v8h*)(ap + hi * 16);
            *((v8h*)&aF[u] + 1) = *(const v8h*)(ap + hi * 16 + 8);
        }
#pragma unroll
        for (int t = 0; t < 4; ++t) {
            const _Float16* bp = BT + (size_t)(n0 + t * 16 + nlo) * Kdim + k0 + hi * 16;
            __builtin_prefetch(bp + 32, 0, 1);   // next k-chunk of B
            v16h bfrag = *(const v16h*)bp;
#pragma unroll
            for (int u = 0; u < 2; ++u)
                acc[u][t] = __builtin_amdgcn_wmma_f32_16x16x32_f16(
                    false, aF[u], false, bfrag, (short)0, acc[u][t], false, false);
        }
        __syncthreads();   // protect sA[buf] before it is refilled next iter
    }

#pragma unroll
    for (int u = 0; u < 2; ++u) {
#pragma unroll
        for (int t = 0; t < 4; ++t) {
            const int n = n0 + t * 16 + nlo;
            const float bv = bias ? bias[n] : 0.0f;
#pragma unroll
            for (int v = 0; v < 8; ++v) {
                const int m = m0 + u * 16 + v + hi * 8;
                float val = acc[u][t][v] + bv;
                if (relu) val = fmaxf(val, 0.0f);
                if (vtrans) {
                    // store V transposed: Vt[b][h][d][s]
                    const int bb = m >> 11, s = m & (TB_S - 1);
                    const int hh = n >> 6,  dd = n & 63;
                    outH[((size_t)((bb * TB_H + hh) * TB_DH + dd)) * TB_S + s] =
                        (_Float16)val;
                } else {
                    if (outH) outH[(size_t)m * Ndim + n] = (_Float16)val;
                    if (outF) outF[(size_t)m * Ndim + n] = val;
                }
            }
        }
    }
}

// ---------------------------------------------------------------------------
// Flash attention: one wave per (b, h, 16-query tile).
// Loop over key tiles of 32: 4 WMMA for scores (16x32), online softmax via LDS,
// 4 WMMA for P @ V (V pre-transposed so B-fragments are contiguous over keys).
// ---------------------------------------------------------------------------
__global__ __launch_bounds__(32) void attn_flash(
    const _Float16* __restrict__ Qh, const _Float16* __restrict__ Kh,
    const _Float16* __restrict__ Vt, _Float16* __restrict__ ctxh) {
    __shared__ alignas(16) float    sS[16][32];
    __shared__ alignas(16) _Float16 sP[16][32];
    __shared__ float sScale[16];
    __shared__ float sInv[16];

    const int lane = threadIdx.x;
    const int nlo  = lane & 15;
    const int hi   = lane >> 4;
    const int qt   = blockIdx.x;   // 0..S/16-1
    const int h    = blockIdx.y;   // 0..H-1
    const int b    = blockIdx.z;   // 0..B-1

    const int q0   = b * TB_S + qt * 16;    // first global query row
    const int colQ = h * TB_DH;

    // Preload Q A-fragments for the two d-halves (d 0..31, 32..63)
    const _Float16* qrow = Qh + (size_t)(q0 + nlo) * TB_D + colQ;
    v16h aQ0, aQ1;
    *(v8h*)&aQ0       = *(const v8h*)(qrow + hi * 16);
    *((v8h*)&aQ0 + 1) = *(const v8h*)(qrow + hi * 16 + 8);
    *(v8h*)&aQ1       = *(const v8h*)(qrow + 32 + hi * 16);
    *((v8h*)&aQ1 + 1) = *(const v8h*)(qrow + 32 + hi * 16 + 8);

    v8f oacc[4] = {};
    float rowMax = -1e30f;
    float rowSum = 0.0f;
    const float scale = 0.125f;   // 1/sqrt(64)

    const _Float16* vbase = Vt + (size_t)((b * TB_H + h) * TB_DH) * TB_S;

    for (int key0 = 0; key0 < TB_S; key0 += 32) {
        // ---- scores: two 16-key chunks ----
#pragma unroll
        for (int c = 0; c < 2; ++c) {
            const _Float16* krow =
                Kh + (size_t)(b * TB_S + key0 + c * 16 + nlo) * TB_D + colQ;
            v16h bK0 = *(const v16h*)(krow + hi * 16);
            v16h bK1 = *(const v16h*)(krow + 32 + hi * 16);
            v8f sc = {};
            sc = __builtin_amdgcn_wmma_f32_16x16x32_f16(
                false, aQ0, false, bK0, (short)0, sc, false, false);
            sc = __builtin_amdgcn_wmma_f32_16x16x32_f16(
                false, aQ1, false, bK1, (short)0, sc, false, false);
#pragma unroll
            for (int v = 0; v < 8; ++v)
                sS[v + hi * 8][c * 16 + nlo] = sc[v] * scale;
        }
        __syncthreads();

        // ---- online softmax, one row per lane<16 ----
        if (lane < 16) {
            const int r = lane;
            float mNew = rowMax;
#pragma unroll
            for (int j = 0; j < 32; ++j) mNew = fmaxf(mNew, sS[r][j]);
            const float f = __expf(rowMax - mNew);
            float l = rowSum * f;
#pragma unroll
            for (int j = 0; j < 32; ++j) {
                const float p = __expf(sS[r][j] - mNew);
                l += p;
                sP[r][j] = (_Float16)p;
            }
            sScale[r] = f;
            rowMax = mNew;
            rowSum = l;
        }
        __syncthreads();

        // ---- rescale accumulators ----
#pragma unroll
        for (int v = 0; v < 8; ++v) {
            const float f = sScale[v + hi * 8];
#pragma unroll
            for (int t = 0; t < 4; ++t) oacc[t][v] *= f;
        }

        // ---- P fragment (A layout) from LDS ----
        v16h aP;
    *(v8h*)&aP       = *(const v8h*)(&sP[nlo][hi * 16]);
    *((v8h*)&aP + 1) = *(const v8h*)(&sP[nlo][hi * 16 + 8]);

        // ---- ctx += P @ V ----
#pragma unroll
        for (int t = 0; t < 4; ++t) {
            const _Float16* vp = vbase + (size_t)(t * 16 + nlo) * TB_S + key0 + hi * 16;
            v16h bV = *(const v16h*)vp;
            oacc[t] = __builtin_amdgcn_wmma_f32_16x16x32_f16(
                false, aP, false, bV, (short)0, oacc[t], false, false);
        }
        __syncthreads();   // protect sS/sP/sScale before next iteration
    }

    if (lane < 16) sInv[lane] = 1.0f / rowSum;
    __syncthreads();

#pragma unroll
    for (int t = 0; t < 4; ++t) {
#pragma unroll
        for (int v = 0; v < 8; ++v) {
            const int row = v + hi * 8;
            const int d   = t * 16 + nlo;
            ctxh[(size_t)(q0 + row) * TB_D + colQ + d] =
                (_Float16)(oacc[t][v] * sInv[row]);
        }
    }
}

// ---------------------------------------------------------------------------
// Residual + LayerNorm over D=512; one block (256 threads) per row.
// Emits f32 (for the next residual) and optional f16 (for the next GEMM).
// ---------------------------------------------------------------------------
__global__ __launch_bounds__(256) void residual_layernorm(
    const float* __restrict__ a, const float* __restrict__ b,
    const float* __restrict__ g, const float* __restrict__ be,
    float* __restrict__ outF, _Float16* __restrict__ outH) {
    __shared__ float red[256];
    const int row = blockIdx.x;
    const int t   = threadIdx.x;
    const float* pa = a + (size_t)row * TB_D;
    const float* pb = b + (size_t)row * TB_D;

    const float v0 = pa[t] + pb[t];
    const float v1 = pa[t + 256] + pb[t + 256];

    red[t] = v0 + v1;
    __syncthreads();
#pragma unroll
    for (int off = 128; off > 0; off >>= 1) {
        if (t < off) red[t] += red[t + off];
        __syncthreads();
    }
    const float mu = red[0] * (1.0f / TB_D);
    __syncthreads();

    const float d0 = v0 - mu, d1 = v1 - mu;
    red[t] = d0 * d0 + d1 * d1;
    __syncthreads();
#pragma unroll
    for (int off = 128; off > 0; off >>= 1) {
        if (t < off) red[t] += red[t + off];
        __syncthreads();
    }
    const float rstd = rsqrtf(red[0] * (1.0f / TB_D) + 1e-5f);

    const float o0 = d0 * rstd * g[t] + be[t];
    const float o1 = d1 * rstd * g[t + 256] + be[t + 256];
    if (outF) {
        outF[(size_t)row * TB_D + t]       = o0;
        outF[(size_t)row * TB_D + t + 256] = o1;
    }
    if (outH) {
        outH[(size_t)row * TB_D + t]       = (_Float16)o0;
        outH[(size_t)row * TB_D + t + 256] = (_Float16)o1;
    }
}

// ---------------------------------------------------------------------------
// Host-side orchestration
// ---------------------------------------------------------------------------
extern "C" void kernel_launch(void* const* d_in, const int* in_sizes, int n_in,
                              void* d_out, int out_size, void* d_ws, size_t ws_size,
                              hipStream_t stream) {
    const float* x  = (const float*)d_in[0];
    const float* Wq = (const float*)d_in[1];
    const float* bq = (const float*)d_in[2];
    const float* Wk = (const float*)d_in[3];
    const float* bk = (const float*)d_in[4];
    const float* Wv = (const float*)d_in[5];
    const float* bv = (const float*)d_in[6];
    const float* Wo = (const float*)d_in[7];
    const float* bo = (const float*)d_in[8];
    const float* W1 = (const float*)d_in[9];
    const float* b1 = (const float*)d_in[10];
    const float* W2 = (const float*)d_in[11];
    const float* b2 = (const float*)d_in[12];
    const float* g1  = (const float*)d_in[13];
    const float* be1 = (const float*)d_in[14];
    const float* g2  = (const float*)d_in[15];
    const float* be2 = (const float*)d_in[16];
    float* out = (float*)d_out;

    // ---- workspace layout (256B aligned) ----
    size_t off = 0;
    auto alloc = [&](size_t bytes) -> void* {
        void* p = (char*)d_ws + off;
        off += (bytes + 255) & ~(size_t)255;
        return p;
    };
    _Float16* xh   = (_Float16*)alloc((size_t)TB_M * TB_D * 2);
    _Float16* WqT  = (_Float16*)alloc((size_t)TB_D * TB_D * 2);
    _Float16* WkT  = (_Float16*)alloc((size_t)TB_D * TB_D * 2);
    _Float16* WvT  = (_Float16*)alloc((size_t)TB_D * TB_D * 2);
    _Float16* WoT  = (_Float16*)alloc((size_t)TB_D * TB_D * 2);
    _Float16* W1T  = (_Float16*)alloc((size_t)TB_DFF * TB_D * 2);
    _Float16* W2T  = (_Float16*)alloc((size_t)TB_D * TB_DFF * 2);
    _Float16* Qh   = (_Float16*)alloc((size_t)TB_M * TB_D * 2);
    _Float16* Kh   = (_Float16*)alloc((size_t)TB_M * TB_D * 2);
    _Float16* Vt   = (_Float16*)alloc((size_t)TB_M * TB_D * 2);
    _Float16* ctxh = (_Float16*)alloc((size_t)TB_M * TB_D * 2);
    float*    tmpF = (float*)alloc((size_t)TB_M * TB_D * 4);
    float*    hF   = (float*)alloc((size_t)TB_M * TB_D * 4);
    _Float16* ff1  = Qh;   // overlay: Q/K/Vt/ctx region (16MB) free after O-proj
    _Float16* hH   = xh;   // overlay: x-f16 free after QKV projections

    // ---- 1. conversions / weight transposes ----
    {
        int n = TB_M * TB_D;
        cvt_f32_to_f16<<<(n + 255) / 256, 256, 0, stream>>>(x, xh, n);
    }
    {
        int n = TB_D * TB_D;
        dim3 g((n + 255) / 256);
        transpose_cvt<<<g, 256, 0, stream>>>(Wq, WqT, TB_D, TB_D);
        transpose_cvt<<<g, 256, 0, stream>>>(Wk, WkT, TB_D, TB_D);
        transpose_cvt<<<g, 256, 0, stream>>>(Wv, WvT, TB_D, TB_D);
        transpose_cvt<<<g, 256, 0, stream>>>(Wo, WoT, TB_D, TB_D);
    }
    {
        int n = TB_D * TB_DFF;
        dim3 g((n + 255) / 256);
        transpose_cvt<<<g, 256, 0, stream>>>(W1, W1T, TB_D, TB_DFF);  // [2048][512]
        transpose_cvt<<<g, 256, 0, stream>>>(W2, W2T, TB_DFF, TB_D);  // [512][2048]
    }

    // ---- 2. QKV projections ----
    {
        dim3 grid(TB_D / 256, TB_M / 32);
        gemm_wmma<<<grid, 128, 0, stream>>>(xh, WqT, bq, nullptr, Qh,
                                            TB_M, TB_D, TB_D, 0, 0);
        gemm_wmma<<<grid, 128, 0, stream>>>(xh, WkT, bk, nullptr, Kh,
                                            TB_M, TB_D, TB_D, 0, 0);
        gemm_wmma<<<grid, 128, 0, stream>>>(xh, WvT, bv, nullptr, Vt,
                                            TB_M, TB_D, TB_D, 0, 1);  // transposed store
    }

    // ---- 3. flash attention ----
    {
        dim3 grid(TB_S / 16, TB_H, TB_B);
        attn_flash<<<grid, 32, 0, stream>>>(Qh, Kh, Vt, ctxh);
    }

    // ---- 4. output projection ----
    {
        dim3 grid(TB_D / 256, TB_M / 32);
        gemm_wmma<<<grid, 128, 0, stream>>>(ctxh, WoT, bo, tmpF, nullptr,
                                            TB_M, TB_D, TB_D, 0, 0);
    }

    // ---- 5. residual + LN 1 ----
    residual_layernorm<<<TB_M, 256, 0, stream>>>(x, tmpF, g1, be1, hF, hH);

    // ---- 6. FFN ----
    {
        dim3 grid1(TB_DFF / 256, TB_M / 32);
        gemm_wmma<<<grid1, 128, 0, stream>>>(hH, W1T, b1, nullptr, ff1,
                                             TB_M, TB_DFF, TB_D, 1, 0);  // ReLU
        dim3 grid2(TB_D / 256, TB_M / 32);
        gemm_wmma<<<grid2, 128, 0, stream>>>(ff1, W2T, b2, tmpF, nullptr,
                                             TB_M, TB_D, TB_DFF, 0, 0);
    }

    // ---- 7. residual + LN 2 -> output ----
    residual_layernorm<<<TB_M, 256, 0, stream>>>(hF, tmpF, g2, be2, out, nullptr);
}
